// DeepLSTM_79448305041980
// MI455X (gfx1250) — matrix operand order repaired and checked by reading, not compiled, not run on hardware
//
#include <hip/hip_runtime.h>
#include <hip/hip_bf16.h>

// CDNA5 / gfx1250 persistent deep-LSTM + MDN head.
// Grid: 16 column-groups (x) * 8 batch-groups (y) = 128 persistent WGs.
// Each WG: 128 threads = 4 waves (wave32). Wave w owns gate block w (i,f,g,o)
// for the WG's 16 hidden columns; M=16 batch rows = one WMMA tile.
// Weights live in LDS as fp16 (~215KB/WGP). h exchanged via L2 (fp16,
// parity double-buffered); 3 global spin barriers per time step among the
// 16 column-groups of a batch group. h-blocks are pulled into LDS with
// CDNA5 global_load_async_to_lds_b128 (ASYNCcnt).

typedef __attribute__((ext_vector_type(16))) _Float16 v16h;
typedef __attribute__((ext_vector_type(8)))  _Float16 h8;
typedef __attribute__((ext_vector_type(4)))  _Float16 h4;
typedef __attribute__((ext_vector_type(8)))  float    v8f;

namespace {
constexpr int Hd   = 256;
constexpr int Kmix = 20;
constexpr int INs  = 2;
constexpr int GOUT = 100;   // OUT*K
constexpr int Bb   = 128;
constexpr int Tt   = 1000;
constexpr int KP0  = 288;   // pad(256+2) to 9 k-tiles of 32
constexpr int KP1  = 544;   // pad(256+256+2) to 17 k-tiles
constexpr int NCG  = 16;    // column groups
constexpr int NBG  = 8;     // batch groups
constexpr int ROWS = 16;    // batch rows per group (one WMMA M-tile)
constexpr int BLK  = 128;   // 4 waves

// Activation column layout (weights repacked to match):
//   layers 1/2 (KP1): [h_{l-1} cur (0..255) | h_l prev (256..511) | x (512..513) | 0]
//   layer 0 / head (KP0): [h (0..255) | x (256..257) | 0]

// ---- LDS layout (bytes, all 16B aligned) ----
constexpr int OFF_W0  = 0;                      constexpr int SZ_W0 = 64 * KP0 * 2;
constexpr int OFF_W1  = OFF_W0 + SZ_W0;         constexpr int SZ_W1 = 64 * KP1 * 2;
constexpr int OFF_W2  = OFF_W1 + SZ_W1;
constexpr int OFF_WH  = OFF_W2 + SZ_W1;         constexpr int SZ_WH = 32 * KP0 * 2;
constexpr int OFF_ACT = OFF_WH + SZ_WH;         constexpr int SZ_ACT = ROWS * KP1 * 2;
constexpr int OFF_DEX = OFF_ACT + SZ_ACT;       constexpr int SZ_DEX = 4 * 16 * 16 * 4;
constexpr int OFF_CST = OFF_DEX + SZ_DEX;       constexpr int SZ_CST = 3 * 16 * 16 * 4;
constexpr int OFF_B0  = OFF_CST + SZ_CST;
constexpr int OFF_B1  = OFF_B0 + 64 * 4;
constexpr int OFF_B2  = OFF_B1 + 64 * 4;
constexpr int OFF_BH  = OFF_B2 + 64 * 4;
constexpr int SMEM_BYTES = OFF_BH + 32 * 4;
static_assert(SMEM_BYTES <= 320 * 1024, "exceeds WGP LDS");

// ---- workspace layout ----
constexpr size_t HBUF_BYTES = 2ull * 3 * Bb * Hd * 2;     // [parity][layer][B][H] fp16
constexpr size_t CTR_COUNT  = (size_t)NBG * 3 * Tt;       // per-phase arrival counters
constexpr size_t WS_NEEDED  = HBUF_BYTES + CTR_COUNT * 4;
} // namespace

__device__ __forceinline__ float sigf(float v) { return 1.0f / (1.0f + __expf(-v)); }

// Flat pointers in the LDS aperture keep the LDS byte address in the low
// 32 bits (ISA 10.2).
__device__ __forceinline__ unsigned lds_off_u32(const void* p) {
  return (unsigned)(uintptr_t)p;
}

// Async copy of 64B (this thread's share of one 256-half h block) from the
// global fp16 h buffer straight into the LDS activation row. Tracked by
// ASYNCcnt; producer waits with s_wait_asynccnt before consuming.
__device__ __forceinline__ void async_h_block(_Float16* __restrict__ actcol,
                                              const _Float16* __restrict__ hrow,
                                              int sub) {
#pragma unroll
  for (int cchunk = 0; cchunk < 4; ++cchunk) {
    const unsigned loff = lds_off_u32(actcol + sub * 32 + cchunk * 8);
    const unsigned long long ga =
        (unsigned long long)(const void*)(hrow + sub * 32 + cchunk * 8);
    asm volatile("global_load_async_to_lds_b128 %0, %1, off"
                 :: "v"(loff), "v"(ga) : "memory");
  }
}

// Build activation rows: h blocks via async-to-LDS, x + pad via one h4 store.
template <int KP>
__device__ __forceinline__ void build_act(_Float16* __restrict__ act,
                                          const float* __restrict__ x, int t,
                                          const _Float16* __restrict__ h1g,
                                          const _Float16* __restrict__ h2g,
                                          int row0) {
  const int tid = threadIdx.x;
  const int r = tid >> 3, sub = tid & 7;       // 16 rows, 8 threads/row
  const size_t grow = (size_t)(row0 + r);
  _Float16* arow = act + (size_t)r * KP;

  async_h_block(arow, h1g + grow * Hd, sub);
  if (KP == KP1) async_h_block(arow + 256, h2g + grow * Hd, sub);

  constexpr int tb = (KP == KP1) ? 512 : 256;  // 32-half tail: [x0,x1,0,...]
  h4 tv = {(_Float16)0.0f, (_Float16)0.0f, (_Float16)0.0f, (_Float16)0.0f};
  if (sub == 0) {
    const float* xp = x + (grow * Tt + t) * INs;
    tv[0] = (_Float16)xp[0];
    tv[1] = (_Float16)xp[1];
  }
  *(h4*)(arow + tb + sub * 4) = tv;

  asm volatile("s_wait_asynccnt 0x0" ::: "memory");
}

// One 16x16 WMMA output tile: A = act rows [16 x KP], B = Wrow0 rows [16 x KP]
// (W stored row-major per output column n; B fragment k-major is contiguous).
__device__ __forceinline__ v8f wmma_tile(const _Float16* __restrict__ Wrow0,
                                         const _Float16* __restrict__ A,
                                         int KP, int ktiles, float binit) {
  const int lane = threadIdx.x & 31;
  const int n    = lane & 15;
  const int hi   = lane >> 4;
  const _Float16* ap = A     + (size_t)n * KP + (hi ? 8  : 0);
  const _Float16* bp = Wrow0 + (size_t)n * KP + (hi ? 8  : 0);
  v8f acc;
#pragma unroll
  for (int e = 0; e < 8; ++e) acc[e] = binit;
  for (int kt = 0; kt < ktiles; ++kt) {
    const int k0 = kt << 5;
    h8 alo = *(const h8*)(ap + k0);
    h8 ahi = *(const h8*)(ap + k0 + 16);
    h8 blo = *(const h8*)(bp + k0);
    h8 bhi = *(const h8*)(bp + k0 + 16);
    v16h av, bv;
#pragma unroll
    for (int e = 0; e < 8; ++e) {
      av[e] = alo[e]; av[8 + e] = ahi[e];
      bv[e] = blo[e]; bv[8 + e] = bhi[e];
    }
    acc = __builtin_amdgcn_wmma_f32_16x16x32_f16(
        false, av, false, bv, (short)0, acc, false, false);
  }
  return acc;
}

__device__ __forceinline__ void lstm_update(const float* __restrict__ dex,
                                            float* __restrict__ cst, int l,
                                            _Float16* __restrict__ hout,
                                            int row0, int col0) {
  const int tid = threadIdx.x;
  const int r = tid >> 3;              // (tid*2)/16
  const int c = (tid * 2) & 15;        // even
  union { _Float16 h[2]; unsigned u32; } pk;
#pragma unroll
  for (int u = 0; u < 2; ++u) {
    const float gi = dex[(0 * 16 + r) * 16 + c + u];
    const float gf = dex[(1 * 16 + r) * 16 + c + u];
    const float gg = dex[(2 * 16 + r) * 16 + c + u];
    const float go = dex[(3 * 16 + r) * 16 + c + u];
    float* cp = cst + (l * 16 + r) * 16 + c + u;
    const float cn = sigf(gf) * (*cp) + sigf(gi) * tanhf(gg);
    *cp = cn;
    pk.h[u] = (_Float16)(sigf(go) * tanhf(cn));
  }
  *(unsigned*)&hout[(size_t)(row0 + r) * Hd + (col0 + c)] = pk.u32;
}

__device__ __forceinline__ void group_barrier(unsigned* __restrict__ ctr) {
  __threadfence();        // release h writes at agent scope
  __syncthreads();
  if (threadIdx.x == 0) {
    __hip_atomic_fetch_add(ctr, 1u, __ATOMIC_RELEASE, __HIP_MEMORY_SCOPE_AGENT);
    while (__hip_atomic_load(ctr, __ATOMIC_ACQUIRE, __HIP_MEMORY_SCOPE_AGENT) <
           (unsigned)NCG)
      __builtin_amdgcn_s_sleep(1);
  }
  __syncthreads();
}

__global__ __launch_bounds__(BLK, 1)
void deep_lstm_persistent(const float* __restrict__ x,
                          const float* __restrict__ Wih0, const float* __restrict__ Whh0,
                          const float* __restrict__ bih0, const float* __restrict__ bhh0,
                          const float* __restrict__ Wih,  const float* __restrict__ Whh,
                          const float* __restrict__ bih,  const float* __restrict__ bhh,
                          const float* __restrict__ Wg,   const float* __restrict__ bg,
                          const float* __restrict__ Ww,   const float* __restrict__ bw,
                          float* __restrict__ gout, float* __restrict__ wout,
                          _Float16* __restrict__ hbuf, unsigned* __restrict__ ctrs) {
  extern __shared__ char smem[];
  _Float16* W0  = (_Float16*)(smem + OFF_W0);
  _Float16* W1  = (_Float16*)(smem + OFF_W1);
  _Float16* W2  = (_Float16*)(smem + OFF_W2);
  _Float16* WHD = (_Float16*)(smem + OFF_WH);
  _Float16* ACT = (_Float16*)(smem + OFF_ACT);
  float*    DEX = (float*)(smem + OFF_DEX);
  float*    CST = (float*)(smem + OFF_CST);
  float*    B0  = (float*)(smem + OFF_B0);
  float*    B1  = (float*)(smem + OFF_B1);
  float*    B2  = (float*)(smem + OFF_B2);
  float*    BH  = (float*)(smem + OFF_BH);

  const int cg   = blockIdx.x;      // column group 0..15 (16 hidden cols each)
  const int bgi  = blockIdx.y;      // batch group 0..7 (16 rows each)
  const int tid  = threadIdx.x;
  const int wave = tid >> 5;
  const int lane = tid & 31;
  const int row0 = bgi * ROWS;
  const int col0 = cg * 16;

  // ---------------- one-time preload: fp32 weights -> fp16 LDS ----------------
  // (column order matches the activation layout described above)
  {
    const int r = tid >> 1, hp = tid & 1;        // 64 rows, 2 threads/row
    const int w = r >> 4, j = r & 15;
    const int N0 = w * 256 + col0 + j;           // global gate row (i,f,g,o blocks)
    for (int k = hp * (KP0 / 2); k < (hp + 1) * (KP0 / 2); ++k) {
      float v = 0.0f;
      if (k < 256)      v = Whh0[(size_t)N0 * 256 + k];
      else if (k < 258) v = Wih0[(size_t)N0 * 2 + (k - 256)];
      W0[(size_t)r * KP0 + k] = (_Float16)v;
    }
    for (int l = 0; l < 2; ++l) {
      _Float16* Wd = l ? W2 : W1;
      const float* wi = Wih + (size_t)l * 1024 * 258;
      const float* wh = Whh + (size_t)l * 1024 * 256;
      for (int k = hp * (KP1 / 2); k < (hp + 1) * (KP1 / 2); ++k) {
        float v = 0.0f;
        if (k < 256)      v = wi[(size_t)N0 * 258 + 2 + k];          // x-skip h part
        else if (k < 512) v = wh[(size_t)N0 * 256 + (k - 256)];      // recurrent part
        else if (k < 514) v = wi[(size_t)N0 * 258 + (k - 512)];      // x part
        Wd[(size_t)r * KP1 + k] = (_Float16)v;
      }
    }
    // head weights: 32 rows, 4 threads/row
    const int hr = tid >> 2, qp = tid & 3;
    for (int k = qp * (KP0 / 4); k < (qp + 1) * (KP0 / 4); ++k) {
      float v = 0.0f;
      if (cg < 7) {
        const int G = col0 + hr;
        if (hr < 16 && G < GOUT) {
          if (k < 256)      v = Wg[(size_t)G * 258 + 2 + k];
          else if (k < 258) v = Wg[(size_t)G * 258 + (k - 256)];
        }
      } else if (cg == 7) {
        if (hr < Kmix) {
          if (k < 256)      v = Ww[(size_t)hr * 258 + 2 + k];
          else if (k < 258) v = Ww[(size_t)hr * 258 + (k - 256)];
        }
      }
      WHD[(size_t)hr * KP0 + k] = (_Float16)v;
    }
    if (tid < 64) {
      const int ww = tid >> 4, jj = tid & 15, N = ww * 256 + col0 + jj;
      B0[tid] = bih0[N] + bhh0[N];
      B1[tid] = bih[N] + bhh[N];
      B2[tid] = bih[1024 + N] + bhh[1024 + N];
    }
    if (tid < 32) {
      float bv = 0.0f;
      if (cg < 7)       { const int G = col0 + tid; if (tid < 16 && G < GOUT) bv = bg[G]; }
      else if (cg == 7) { if (tid < Kmix) bv = bw[tid]; }
      BH[tid] = bv;
    }
  }
  for (int i = tid; i < 3 * 16 * 16; i += BLK) CST[i] = 0.0f;
  __syncthreads();

  auto HB = [&](int par, int l) -> _Float16* {
    return hbuf + (size_t)(par * 3 + l) * Bb * Hd;
  };

  // ---------------- persistent time loop ----------------
  for (int t = 0; t < Tt; ++t) {
    const int p = t & 1, q = p ^ 1;
    unsigned* cbase = ctrs + (size_t)bgi * (3 * Tt) + (size_t)t * 3;

    if (wave == 0 && lane < ROWS) {
      const int tn = (t + 1 < Tt) ? t + 1 : t;
      __builtin_prefetch(&x[((size_t)(row0 + lane) * Tt + tn) * INs], 0, 1);
    }

    // ---- layer 0: gates = [h0_prev | x] @ Wcat0^T + b ----
    build_act<KP0>(ACT, x, t, HB(q, 0), nullptr, row0);
    __syncthreads();
    {
      v8f acc = wmma_tile(W0 + (size_t)wave * 16 * KP0, ACT, KP0, KP0 / 32,
                          B0[wave * 16 + (lane & 15)]);
#pragma unroll
      for (int e = 0; e < 8; ++e)
        DEX[(wave * 16 + e + ((lane >> 4) << 3)) * 16 + (lane & 15)] = acc[e];
    }
    __syncthreads();
    lstm_update(DEX, CST, 0, HB(p, 0), row0, col0);
    group_barrier(cbase + 0);

    // ---- layer 1: gates = [h0_cur | h1_prev | x] @ Wcat1^T + b ----
    build_act<KP1>(ACT, x, t, HB(p, 0), HB(q, 1), row0);
    __syncthreads();
    {
      v8f acc = wmma_tile(W1 + (size_t)wave * 16 * KP1, ACT, KP1, KP1 / 32,
                          B1[wave * 16 + (lane & 15)]);
#pragma unroll
      for (int e = 0; e < 8; ++e)
        DEX[(wave * 16 + e + ((lane >> 4) << 3)) * 16 + (lane & 15)] = acc[e];
    }
    __syncthreads();
    lstm_update(DEX, CST, 1, HB(p, 1), row0, col0);
    group_barrier(cbase + 1);

    // ---- layer 2 ----
    build_act<KP1>(ACT, x, t, HB(p, 1), HB(q, 2), row0);
    __syncthreads();
    {
      v8f acc = wmma_tile(W2 + (size_t)wave * 16 * KP1, ACT, KP1, KP1 / 32,
                          B2[wave * 16 + (lane & 15)]);
#pragma unroll
      for (int e = 0; e < 8; ++e)
        DEX[(wave * 16 + e + ((lane >> 4) << 3)) * 16 + (lane & 15)] = acc[e];
    }
    __syncthreads();
    lstm_update(DEX, CST, 2, HB(p, 2), row0, col0);
    group_barrier(cbase + 2);

    // ---- heads (cg 0..6: g columns; cg 7: mixture weights + softmax) ----
    if (cg < 8) {
      build_act<KP0>(ACT, x, t, HB(p, 2), nullptr, row0);
      __syncthreads();
      if (cg < 7) {
        if (wave == 0) {
          v8f acc = wmma_tile(WHD, ACT, KP0, KP0 / 32, BH[lane & 15]);
          const int col = col0 + (lane & 15);
          if (col < GOUT) {
#pragma unroll
            for (int e = 0; e < 8; ++e) {
              const int r = e + ((lane >> 4) << 3);
              gout[((size_t)(row0 + r) * Tt + t) * GOUT + col] = acc[e];
            }
          }
        }
      } else {
        if (wave < 2) {
          v8f acc = wmma_tile(WHD + (size_t)wave * 16 * KP0, ACT, KP0, KP0 / 32,
                              BH[wave * 16 + (lane & 15)]);
#pragma unroll
          for (int e = 0; e < 8; ++e)
            DEX[(wave * 16 + e + ((lane >> 4) << 3)) * 16 + (lane & 15)] = acc[e];
        }
        __syncthreads();
        if (tid < ROWS) {
          float v[Kmix];
#pragma unroll
          for (int kk = 0; kk < Kmix; ++kk)
            v[kk] = (kk < 16) ? DEX[tid * 16 + kk]
                              : DEX[(16 + tid) * 16 + (kk - 16)];
          float m = v[0];
          for (int kk = 1; kk < Kmix; ++kk) m = fmaxf(m, v[kk]);
          float s = 0.0f;
          for (int kk = 0; kk < Kmix; ++kk) { v[kk] = __expf(v[kk] - m); s += v[kk]; }
          const float inv = 1.0f / s;
          for (int kk = 0; kk < Kmix; ++kk)
            wout[((size_t)(row0 + tid) * Tt + t) * Kmix + kk] = v[kk] * inv;
        }
      }
      __syncthreads();
    }
  }
}

extern "C" void kernel_launch(void* const* d_in, const int* in_sizes, int n_in,
                              void* d_out, int out_size, void* d_ws, size_t ws_size,
                              hipStream_t stream) {
  (void)in_sizes; (void)n_in; (void)out_size; (void)ws_size;
  const float* x    = (const float*)d_in[0];
  const float* Wih0 = (const float*)d_in[1];
  const float* Whh0 = (const float*)d_in[2];
  const float* bih0 = (const float*)d_in[3];
  const float* bhh0 = (const float*)d_in[4];
  const float* Wih  = (const float*)d_in[5];
  const float* Whh  = (const float*)d_in[6];
  const float* bih  = (const float*)d_in[7];
  const float* bhh  = (const float*)d_in[8];
  const float* Wg   = (const float*)d_in[9];
  const float* bg   = (const float*)d_in[10];
  const float* Ww   = (const float*)d_in[11];
  const float* bw   = (const float*)d_in[12];

  float* gout = (float*)d_out;                       // [B,T,100]
  float* wout = gout + (size_t)Bb * Tt * GOUT;       // [B,T,20]

  _Float16* hbuf = (_Float16*)d_ws;                  // [2][3][B][H] fp16
  unsigned* ctrs = (unsigned*)((char*)d_ws + HBUF_BYTES);

  // Zero h state + barrier counters (graph-capturable, deterministic per call).
  hipMemsetAsync(d_ws, 0, WS_NEEDED, stream);

  hipFuncSetAttribute((const void*)deep_lstm_persistent,
                      hipFuncAttributeMaxDynamicSharedMemorySize, SMEM_BYTES);

  deep_lstm_persistent<<<dim3(NCG, NBG, 1), dim3(BLK, 1, 1), SMEM_BYTES, stream>>>(
      x, Wih0, Whh0, bih0, bhh0, Wih, Whh, bih, bhh, Wg, bg, Ww, bw,
      gout, wout, hbuf, ctrs);
}
